// StructuredStateSpaceSequenceModel_64871186038989
// MI455X (gfx1250) — compile-verified, matrix operand order checked by reading, not run on hardware
//
#include <hip/hip_runtime.h>
#include <hip/hip_bf16.h>

// ---------------- types ----------------
typedef __bf16 bf16_t;
typedef __attribute__((ext_vector_type(16))) __bf16 v16bf;
typedef __attribute__((ext_vector_type(8)))  __bf16 v8bf;
typedef __attribute__((ext_vector_type(8)))  float  v8f;

// ---------------- model dims ----------------
constexpr int B_ = 16, L_ = 2048, D_ = 256, N_ = 32, V_ = 1024;
constexpr int NLAYERS_ = 4, NS4_ = 8, NFF_ = 4;
constexpr int M_ = B_ * L_;              // 32768 rows of [*, D]

#define DEVFN static __device__ __forceinline__

DEVFN float gelu_f(float x) {
  float x3 = x * x * x;
  return 0.5f * x * (1.f + tanhf(0.7978845608028654f * (x + 0.044715f * x3)));
}
DEVFN float sigmoid_f(float x) { return 1.f / (1.f + __expf(-x)); }

DEVFN bf16_t to_bf16(float f) {
  union { float f; unsigned u; } c; c.f = f;
  unsigned r = (c.u + 0x7FFFu + ((c.u >> 16) & 1u)) >> 16;   // RNE
  union { unsigned short s; bf16_t h; } o; o.s = (unsigned short)r;
  return o.h;
}

// A/B fragment per CDNA5 16-bit 16x32 layout:
// lane m = lane&15 (row for A / col for B^T), kb = (lane>>4)*8.
// elements 0..7  = K kb+0..7   (VGPR0..3)
// elements 8..15 = K kb+16..23 (VGPR4..7)
DEVFN v16bf load_frag(const bf16_t* rowp, int k) {
  v8bf lo = *(const v8bf*)(rowp + k);
  v8bf hi = *(const v8bf*)(rowp + k + 16);
  v16bf r;
#pragma unroll
  for (int i = 0; i < 8; ++i) { r[i] = lo[i]; r[i + 8] = hi[i]; }
  return r;
}

DEVFN v8f wmma_bf16(v16bf a, v16bf b, v8f c) {
  return __builtin_amdgcn_wmma_f32_16x16x32_bf16(false, a, false, b, (short)0, c,
                                                 false, false);
}

// Async DMA of one 16B beat from global into LDS (gfx1250, ASYNCcnt-tracked).
DEVFN void async_copy_b128(unsigned lds_off, const void* gaddr) {
  asm volatile("global_load_async_to_lds_b128 %0, %1, off"
               :: "v"(lds_off), "v"((unsigned long long)(size_t)gaddr)
               : "memory");
}
DEVFN void wait_async0() {
  asm volatile("s_wait_asynccnt 0x0" ::: "memory");
}

// ---------------- weight convert + transpose: W[K,N] f32 -> Wt[N,K] bf16 ----
__global__ void s4ssm_cvt_t_k(const float* __restrict__ W, bf16_t* __restrict__ Wt,
                              int K, int Nc) {
  size_t i = (size_t)blockIdx.x * blockDim.x + threadIdx.x;
  if (i >= (size_t)K * Nc) return;
  int k = (int)(i / Nc);
  int n = (int)(i - (size_t)k * Nc);
  Wt[(size_t)n * K + k] = to_bf16(W[i]);
}

// ---------------- embedding gather ----------------
__global__ void s4ssm_embed_k(const int* __restrict__ batch,
                              const float* __restrict__ emb,
                              float* __restrict__ x) {
  size_t i = (size_t)blockIdx.x * blockDim.x + threadIdx.x;
  if (i >= (size_t)M_ * D_) return;
  int row = (int)(i >> 8);        // / D_ (256)
  int d   = (int)(i & 255);
  x[i] = emb[(size_t)batch[row] * D_ + d];
}

// ---------------- layernorm (one 256-thread block per row) ----------------
template <int WF32, int WBF>
__global__ __launch_bounds__(256) void s4ssm_ln_k(const float* __restrict__ x,
                                                  const float* __restrict__ w,
                                                  const float* __restrict__ b,
                                                  float* __restrict__ uo,
                                                  bf16_t* __restrict__ ho) {
  const int row = blockIdx.x, t = threadIdx.x;
  float v = x[(size_t)row * D_ + t];
  float s = v, q = v * v;
#pragma unroll
  for (int o = 16; o; o >>= 1) { s += __shfl_xor(s, o, 32); q += __shfl_xor(q, o, 32); }
  __shared__ float ps[8], pq[8];
  __shared__ float mu_s, r_s;
  if ((t & 31) == 0) { ps[t >> 5] = s; pq[t >> 5] = q; }
  __syncthreads();
  if (t == 0) {
    float S = 0.f, Q = 0.f;
#pragma unroll
    for (int i = 0; i < 8; ++i) { S += ps[i]; Q += pq[i]; }
    float mu = S * (1.0f / D_);
    float var = Q * (1.0f / D_) - mu * mu;
    mu_s = mu;
    r_s = rsqrtf(var + 1e-5f);
  }
  __syncthreads();
  float y = (v - mu_s) * r_s * w[t] + b[t];
  if (WF32) uo[(size_t)row * D_ + t] = y;
  if (WBF) ho[(size_t)row * D_ + t] = to_bf16(y);
}

// ---------------- S4D diagonal SSM scan (wave32: one complex state per lane) --
__global__ __launch_bounds__(256) void s4ssm_scan_k(
    const float* __restrict__ u, const float* __restrict__ log_dt,
    const float* __restrict__ Alr, const float* __restrict__ Aim,
    const float* __restrict__ Cre, const float* __restrict__ Cim,
    const float* __restrict__ Dsk, bf16_t* __restrict__ g) {
  const int lane = threadIdx.x & 31;
  const int wv = blockIdx.x * 8 + (threadIdx.x >> 5);   // wave id = b*D + d
  const int b = wv / D_, d = wv - (wv / D_) * D_;
  const float dt = __expf(log_dt[d]);
  const float ar = -__expf(Alr[d * N_ + lane]);
  const float ai = Aim[d * N_ + lane];
  const float e = __expf(dt * ar);
  float sy, cy;
  __sincosf(dt * ai, &sy, &cy);
  const float wr = e * cy, wi = e * sy;                 // w = exp(dt*A)
  const float nr = wr - 1.f, ni = wi;                   // (w-1)/A
  const float inv = 1.f / (ar * ar + ai * ai);
  const float fr = (nr * ar + ni * ai) * inv;
  const float fi = (ni * ar - nr * ai) * inv;
  const float cr = Cre[d * N_ + lane], ci = Cim[d * N_ + lane];
  const float Cdr = cr * fr - ci * fi;
  const float Cdi = cr * fi + ci * fr;
  const float dsk = Dsk[d];
  float sr = 0.f, si = 0.f;
  const size_t base = (size_t)b * L_;
  for (int t0 = 0; t0 < L_; t0 += 32) {
    float uv = u[(base + t0 + lane) * D_ + d];
    float keep = 0.f;
#pragma unroll 4
    for (int j = 0; j < 32; ++j) {
      float ut = __shfl(uv, j, 32);
      float nsr = fmaf(wr, sr, fmaf(-wi, si, ut));
      float nsi = fmaf(wr, si, wi * sr);
      sr = nsr; si = nsi;
      float c = 2.f * (Cdr * sr - Cdi * si);
#pragma unroll
      for (int o = 16; o; o >>= 1) c += __shfl_xor(c, o, 32);
      float y = gelu_f(c + ut * dsk);
      if (j == lane) keep = y;
    }
    g[(base + t0 + lane) * D_ + d] = to_bf16(keep);
  }
}

// ---------------- WMMA bf16 GEMM, async-LDS A-panel, fused epilogues ---------
// EPI 0: GLU + residual:  x[M,N] += z[:, :N] * sigmoid(z[:, N:2N])   (Bt has 2N rows)
// EPI 1: gelu -> bf16:    outh[M,N] = bf16(gelu(acc + bias))
// EPI 2: residual add:    outf[M,N] += acc + bias
// EPI 3: bias only:       outf[M,N]  = acc + bias  (nontemporal, streaming out)
// Each wave: 16 rows x (4 tiles of 16 cols). Bt is [Ncols, K] (pre-transposed).
// The block's shared A panel (16 or 32 rows x K) is DMA'd into LDS with
// global_load_async_to_lds_b128 and read back as ds_load_b128 fragments.
extern __shared__ char s4ssm_smem[];

template <int EPI>
__global__ __launch_bounds__(256) void s4ssm_gemm_k(
    const bf16_t* __restrict__ A, const bf16_t* __restrict__ Bt,
    const float* __restrict__ bias, float* __restrict__ outf,
    bf16_t* __restrict__ outh, int K, int N) {
  bf16_t* atile = (bf16_t*)s4ssm_smem;
  const int lane = threadIdx.x & 31;
  const int wid = threadIdx.x >> 5;
  const int strips = (EPI == 0) ? (N >> 5) : (N >> 6);
  const int wave = blockIdx.x * 8 + wid;
  const int rs = wave / strips;
  const int si = wave - rs * strips;
  const int rb = rs * 16;
  const int rb_block = ((blockIdx.x * 8) / strips) * 16;   // block's first row
  const int rows = (strips < 8) ? 16 * (8 / strips) : 16;  // rows in LDS panel

  // ---- cooperative async DMA: A[rb_block .. rb_block+rows) x K -> LDS ----
  {
    const int tileBytes = rows * K * 2;
    const char* gsrc = (const char*)(A + (size_t)rb_block * K);
    const unsigned lds_base = (unsigned)(size_t)s4ssm_smem;
    for (int off = threadIdx.x * 16; off < tileBytes; off += 256 * 16)
      async_copy_b128(lds_base + off, gsrc + off);
    wait_async0();
  }
  __syncthreads();

  int col[4];
  if (EPI == 0) {
    int cb = si * 32;
    col[0] = cb; col[1] = cb + 16; col[2] = cb + N; col[3] = cb + N + 16;
  } else {
    int cb = si * 64;
    col[0] = cb; col[1] = cb + 16; col[2] = cb + 32; col[3] = cb + 48;
  }
  const int mr = lane & 15;
  const int kb = (lane >> 4) << 3;
  const bf16_t* arow = atile + (size_t)((rb - rb_block) + mr) * K;  // LDS
  const bf16_t* brow[4];
#pragma unroll
  for (int j = 0; j < 4; ++j) brow[j] = Bt + (size_t)(col[j] + mr) * K;
  v8f acc[4] = {};
  for (int k = 0; k < K; k += 32) {
    __builtin_prefetch(brow[0] + k + 256, 0, 3);   // keep B strips near
    __builtin_prefetch(brow[2] + k + 256, 0, 3);
    const int kk = k + kb;
    v16bf af = load_frag(arow, kk);                // ds_load_b128 x2
    v16bf b0 = load_frag(brow[0], kk);
    v16bf b1 = load_frag(brow[1], kk);
    v16bf b2 = load_frag(brow[2], kk);
    v16bf b3 = load_frag(brow[3], kk);
    acc[0] = wmma_bf16(af, b0, acc[0]);
    acc[1] = wmma_bf16(af, b1, acc[1]);
    acc[2] = wmma_bf16(af, b2, acc[2]);
    acc[3] = wmma_bf16(af, b3, acc[3]);
  }
  // C/D layout: lane, VGPR v -> row = 8*(lane>>4)+v, col = lane&15
  const int rloc = (lane >> 4) << 3;
#pragma unroll
  for (int v = 0; v < 8; ++v) {
    const int row = rb + rloc + v;
    if (EPI == 0) {
#pragma unroll
      for (int p = 0; p < 2; ++p) {
        float zl = acc[p][v] + bias[col[p] + mr];
        float zr = acc[p + 2][v] + bias[col[p + 2] + mr];
        outf[(size_t)row * N + col[p] + mr] += zl * sigmoid_f(zr);
      }
    } else {
#pragma unroll
      for (int j = 0; j < 4; ++j) {
        const int c = col[j] + mr;
        float val = acc[j][v] + bias[c];
        if (EPI == 1)      outh[(size_t)row * N + c] = to_bf16(gelu_f(val));
        else if (EPI == 2) outf[(size_t)row * N + c] += val;
        else               __builtin_nontemporal_store(val, &outf[(size_t)row * N + c]);
      }
    }
  }
}

// ---------------- host ----------------
extern "C" void kernel_launch(void* const* d_in, const int* in_sizes, int n_in,
                              void* d_out, int out_size, void* d_ws, size_t ws_size,
                              hipStream_t stream) {
  (void)in_sizes; (void)n_in; (void)out_size; (void)ws_size;
  const int*   batch    = (const int*)  d_in[0];
  const float* emb      = (const float*)d_in[1];
  const float* s4_ln_w  = (const float*)d_in[2];
  const float* s4_ln_b  = (const float*)d_in[3];
  const float* s4_logdt = (const float*)d_in[4];
  const float* s4_Alr   = (const float*)d_in[5];
  const float* s4_Aim   = (const float*)d_in[6];
  const float* s4_Cre   = (const float*)d_in[7];
  const float* s4_Cim   = (const float*)d_in[8];
  const float* s4_Dsk   = (const float*)d_in[9];
  const float* s4_Wout  = (const float*)d_in[10];
  const float* s4_bout  = (const float*)d_in[11];
  const float* ff_ln_w  = (const float*)d_in[12];
  const float* ff_ln_b  = (const float*)d_in[13];
  const float* ff_W1    = (const float*)d_in[14];
  const float* ff_b1    = (const float*)d_in[15];
  const float* ff_W2    = (const float*)d_in[16];
  const float* ff_b2    = (const float*)d_in[17];
  const float* fin_ln_w = (const float*)d_in[18];
  const float* fin_ln_b = (const float*)d_in[19];
  const float* W_vocab  = (const float*)d_in[20];
  const float* b_vocab  = (const float*)d_in[21];

  char* wp = (char*)d_ws;
  auto take = [&](size_t bytes) {
    char* p = wp;
    wp += (bytes + 255) & ~(size_t)255;
    return p;
  };
  float*  x     = (float*) take((size_t)M_ * D_ * 4);
  float*  uu    = (float*) take((size_t)M_ * D_ * 4);
  bf16_t* g     = (bf16_t*)take((size_t)M_ * D_ * 2);
  bf16_t* hh    = (bf16_t*)take((size_t)M_ * 4 * D_ * 2);
  bf16_t* WoutT = (bf16_t*)take((size_t)NS4_ * 2 * D_ * D_ * 2);
  bf16_t* W1T   = (bf16_t*)take((size_t)NFF_ * 4 * D_ * D_ * 2);
  bf16_t* W2T   = (bf16_t*)take((size_t)NFF_ * 4 * D_ * D_ * 2);
  bf16_t* WvT   = (bf16_t*)take((size_t)D_ * V_ * 2);

  auto cvt = [&](const float* W, bf16_t* Wt, int K, int Nc) {
    size_t n = (size_t)K * Nc;
    s4ssm_cvt_t_k<<<(unsigned)((n + 255) / 256), 256, 0, stream>>>(W, Wt, K, Nc);
  };
  for (int j = 0; j < NS4_; ++j)
    cvt(s4_Wout + (size_t)j * D_ * 2 * D_, WoutT + (size_t)j * 2 * D_ * D_, D_, 2 * D_);
  for (int l = 0; l < NFF_; ++l)
    cvt(ff_W1 + (size_t)l * D_ * 4 * D_, W1T + (size_t)l * 4 * D_ * D_, D_, 4 * D_);
  for (int l = 0; l < NFF_; ++l)
    cvt(ff_W2 + (size_t)l * 4 * D_ * D_, W2T + (size_t)l * D_ * 4 * D_, 4 * D_, D_);
  cvt(W_vocab, WvT, D_, V_);

  s4ssm_embed_k<<<(M_ * D_) / 256, 256, 0, stream>>>(batch, emb, x);

  auto gemm = [&](int epi, const bf16_t* A, const bf16_t* Bt, const float* bias,
                  float* outf, bf16_t* outh, int K, int N) {
    int strips = (epi == 0) ? N / 32 : N / 64;
    int blocks = ((M_ / 16) * strips) / 8;
    int rows = (strips < 8) ? 16 * (8 / strips) : 16;
    size_t sh = (size_t)rows * K * 2;               // LDS A panel bytes
    switch (epi) {
      case 0: s4ssm_gemm_k<0><<<blocks, 256, sh, stream>>>(A, Bt, bias, outf, outh, K, N); break;
      case 1: s4ssm_gemm_k<1><<<blocks, 256, sh, stream>>>(A, Bt, bias, outf, outh, K, N); break;
      case 2: s4ssm_gemm_k<2><<<blocks, 256, sh, stream>>>(A, Bt, bias, outf, outh, K, N); break;
      default: s4ssm_gemm_k<3><<<blocks, 256, sh, stream>>>(A, Bt, bias, outf, outh, K, N); break;
    }
  };

  for (int l = 0; l < NLAYERS_; ++l) {
    for (int jj = 0; jj < 2; ++jj) {
      const int j = 2 * l + jj;
      s4ssm_ln_k<1, 0><<<M_, 256, 0, stream>>>(x, s4_ln_w + j * D_, s4_ln_b + j * D_,
                                               uu, nullptr);
      s4ssm_scan_k<<<(B_ * D_) / 8, 256, 0, stream>>>(
          uu, s4_logdt + j * D_, s4_Alr + (size_t)j * D_ * N_,
          s4_Aim + (size_t)j * D_ * N_, s4_Cre + (size_t)j * D_ * N_,
          s4_Cim + (size_t)j * D_ * N_, s4_Dsk + j * D_, g);
      gemm(0, g, WoutT + (size_t)j * 2 * D_ * D_, s4_bout + (size_t)j * 2 * D_,
           x, nullptr, D_, D_);
    }
    s4ssm_ln_k<0, 1><<<M_, 256, 0, stream>>>(x, ff_ln_w + l * D_, ff_ln_b + l * D_,
                                             nullptr, g);
    gemm(1, g, W1T + (size_t)l * 4 * D_ * D_, ff_b1 + (size_t)l * 4 * D_,
         nullptr, hh, D_, 4 * D_);
    gemm(2, hh, W2T + (size_t)l * D_ * 4 * D_, ff_b2 + (size_t)l * D_,
         x, nullptr, 4 * D_, D_);
  }
  s4ssm_ln_k<0, 1><<<M_, 256, 0, stream>>>(x, fin_ln_w, fin_ln_b, nullptr, g);
  gemm(3, g, WvT, b_vocab, (float*)d_out, nullptr, D_, V_);
}
